// TransformerEncoder_46411416600666
// MI455X (gfx1250) — matrix operand order repaired and checked by reading, not compiled, not run on hardware
//
#include <hip/hip_runtime.h>
#include <cstdint>
#include <cstddef>

// ---------------- CDNA5 WMMA types ----------------
typedef __bf16 v16bf __attribute__((ext_vector_type(16)));
typedef float  v8f   __attribute__((ext_vector_type(8)));

// ---------------- problem constants ----------------
#define B_    64
#define L_    32
#define M_    16
#define V_    512
#define E_    64
#define D_    192            // d_model = 3*E
#define NH_   4
#define HD_   48
#define NL_   2
#define S_    513            // V + 1 (id token)
#define SP_   544            // padded seq len (34*16, 17*32)
#define ROWS_ (B_*SP_)       // 34816 rows, /16 = 2176 tiles
#define QSCALE 0.14433756729740643f   // 1/sqrt(48)

__device__ __forceinline__ unsigned short f2bf(float f) {
  unsigned int u = __float_as_uint(f);
  u += 0x7FFFu + ((u >> 16) & 1u);          // round-to-nearest-even
  return (unsigned short)(u >> 16);
}

__device__ __forceinline__ v8f wmma_bf16(v16bf a, v16bf b, v8f c) {
  // D = A(16x32 bf16) * B(32x16 bf16) + C(16x16 f32)
  return __builtin_amdgcn_wmma_f32_16x16x32_bf16(false, a, false, b, (short)0, c, false, false);
}

union Frag { v16bf v; uint4 u[2]; unsigned short s[16]; };

// ---------------- elementwise f32 -> bf16 ----------------
__global__ void f2bf_kernel(const float* __restrict__ in, unsigned short* __restrict__ out, int n) {
  int i = blockIdx.x * 256 + threadIdx.x;
  if (i < n) out[i] = f2bf(in[i]);
}

// ---------------- build X: id token, cat embedding, zeros ----------------
__global__ void init_x_kernel(const int* __restrict__ id_arr, const float* __restrict__ id_tab,
                              const float* __restrict__ cat_tab, float* __restrict__ X) {
  int idx = blockIdx.x * 256 + threadIdx.x;   // over ROWS_*D_
  int d = idx % D_;
  int s = (idx / D_) % SP_;
  int b = idx / (D_ * SP_);
  float v = 0.f;
  if (s == 0)                 v = id_tab[(size_t)id_arr[b] * D_ + d];
  else if (s <= V_ && d < E_) v = cat_tab[(s - 1) * E_ + d];
  X[idx] = v;
}

// ---------------- scatter-add dt/pos and amount embeddings ----------------
__global__ void scatter_kernel(const int* __restrict__ cat, const int* __restrict__ dt,
                               const int* __restrict__ amt, const float* __restrict__ dt_tab,
                               const float* __restrict__ pos_tab, const float* __restrict__ amount_tab,
                               float* __restrict__ X) {
  int bl = blockIdx.x;            // b*L + l
  int b = bl / L_, l = bl % L_;
  int e = threadIdx.x;            // 0..63
  float dtpos = dt_tab[dt[bl] * E_ + e] + pos_tab[l * E_ + e];
  for (int m = 0; m < M_; ++m) {
    int c = cat[bl * M_ + m];
    int a = amt[bl * M_ + m];
    float* rowp = X + (size_t)(b * SP_ + 1 + c) * D_;
    atomicAdd(rowp + E_ + e, dtpos);
    atomicAdd(rowp + 2 * E_ + e, amount_tab[a * E_ + e]);
  }
}

// ---------------- WMMA GEMM: Y[M,N] = A[M,192]bf16 @ W[N,192]^T + epilogue ------
// Block = 8 waves = 8 consecutive M-tiles sharing one 64-wide N strip.
// The 64x192 bf16 weight strip is staged global->LDS with the CDNA5 async path
// (GLOBAL_LOAD_ASYNC_TO_LDS_B128, ASYNCcnt), padded rows (200 bf16 = 100 dwords)
// so the 16 fragment-reading lanes hit distinct banks (l16*36 mod 64 all distinct).
enum { EPI_QKV = 0, EPI_RES = 1, EPI_RELU_BF = 2 };

template<int MODE, int N>
__global__ __launch_bounds__(256) void gemm_kernel(
    const unsigned short* __restrict__ A,     // [ROWS_, 192] bf16
    const unsigned short* __restrict__ W,     // [N, 192] bf16
    const float* __restrict__ bias,           // [N]
    const float* __restrict__ resid,          // MODE==EPI_RES
    float* __restrict__ outF,                 // MODE==EPI_RES
    unsigned short* __restrict__ outB,        // MODE==EPI_RELU_BF
    unsigned short* __restrict__ qb,          // MODE==EPI_QKV: [B,NH,SP,64]
    unsigned short* __restrict__ kb,          //                 [B,NH,SP,64]
    unsigned short* __restrict__ vtb) {       //                 [B,NH,64,SP] (transposed)
  constexpr int TN4 = N / 64;
  __shared__ __align__(16) unsigned short sW[64][200];  // 64 W rows, padded stride

  int bn = blockIdx.x % TN4;        // which 64-col strip
  int bm = blockIdx.x / TN4;        // which group of 8 M-tiles

  // ---- async stage of the W strip into LDS: 1536 x 16B chunks, 6 per thread ----
  {
    const unsigned short* Wn = W + (size_t)bn * 64 * D_;
    unsigned int ldsbase = (unsigned int)(uintptr_t)(&sW[0][0]);
    int tid = threadIdx.x;
#pragma unroll
    for (int it = 0; it < 6; ++it) {
      int idx = tid + it * 256;
      int row = idx / 24;           // 0..63  (24 chunks of 8 bf16 per 192-elem row)
      int c   = idx % 24;
      const unsigned short* gp = Wn + row * D_ + c * 8;
      unsigned int lo = ldsbase + row * 400 + c * 16;
      asm volatile("global_load_async_to_lds_b128 %0, %1, off"
                   :: "v"(lo), "v"(gp) : "memory");
    }
    asm volatile("s_wait_asynccnt 0x0" ::: "memory");
    __syncthreads();
  }

  int warp = threadIdx.x >> 5;
  int lane = threadIdx.x & 31;
  int l16 = lane & 15, hi = lane >> 4;
  int tm = bm * 8 + warp;

  const unsigned short* arow = A + (size_t)(tm * 16 + l16) * D_;
  v8f zero = {0.f,0.f,0.f,0.f,0.f,0.f,0.f,0.f};
  v8f acc[4] = {zero, zero, zero, zero};

  for (int ks = 0; ks < D_ / 32; ++ks) {
    int kb0 = ks * 32;
    Frag av;   // A slots: k = j + 16*hi  -> two 16B runs
    av.u[0] = *(const uint4*)(const void*)(arow + kb0 + hi * 16);
    av.u[1] = *(const uint4*)(const void*)(arow + kb0 + hi * 16 + 8);
#pragma unroll
    for (int nt = 0; nt < 4; ++nt) {
      const unsigned short* wrow = &sW[nt * 16 + l16][kb0 + hi * 16]; // B slots: k = j + 16*hi
      Frag bv;
      bv.u[0] = ((const uint4*)(const void*)wrow)[0];
      bv.u[1] = ((const uint4*)(const void*)wrow)[1];
      acc[nt] = wmma_bf16(av.v, bv.v, acc[nt]);
    }
  }

  int rbase = tm * 16 + hi * 8;                 // C/D layout: row = r + 8*hi, col = l16
#pragma unroll
  for (int nt = 0; nt < 4; ++nt) {
    int n = bn * 64 + nt * 16 + l16;
    float bi = bias[n];
#pragma unroll
    for (int r = 0; r < 8; ++r) {
      int grow = rbase + r;
      float v = acc[nt][r] + bi;
      if constexpr (MODE == EPI_RES) {
        outF[(size_t)grow * D_ + n] = v + resid[(size_t)grow * D_ + n];
      } else if constexpr (MODE == EPI_RELU_BF) {
        outB[(size_t)grow * D_ + n] = f2bf(v > 0.f ? v : 0.f);
      } else {  // EPI_QKV: scatter into attention layouts
        int b = grow / SP_, s = grow % SP_;
        int which = n / D_;           // 0=q 1=k 2=v
        int nn = n - which * D_;
        int h = nn / HD_, d = nn % HD_;
        size_t bh = (size_t)(b * NH_ + h);
        if (which == 0)      qb[(bh * SP_ + s) * 64 + d] = f2bf(v * QSCALE);
        else if (which == 1) kb[(bh * SP_ + s) * 64 + d] = f2bf(v);
        else                 vtb[(bh * 64 + d) * SP_ + s] = f2bf(v);
      }
    }
  }
}

// ---------------- flash-attention: one wave per (b,h,16-query tile) ----------------
__global__ __launch_bounds__(256) void attn_kernel(
    const unsigned short* __restrict__ Qb,   // [B*NH, SP, 64] bf16, pre-scaled
    const unsigned short* __restrict__ Kb,   // [B*NH, SP, 64] bf16
    const unsigned short* __restrict__ Vtb,  // [B*NH, 64, SP] bf16
    unsigned short* __restrict__ Ob) {       // [ROWS_, 192] bf16
  __shared__ __align__(16) float sS[8][16][40];   // per-wave score transpose buffer
  int warp = threadIdx.x >> 5, lane = threadIdx.x & 31;
  int l16 = lane & 15, hi = lane >> 4;
  int wid = blockIdx.x * 8 + warp;
  int qt = wid % (SP_ / 16);
  int bh = wid / (SP_ / 16);
  const unsigned short* Qp = Qb + (size_t)bh * SP_ * 64;
  const unsigned short* Kp = Kb + (size_t)bh * SP_ * 64;
  const unsigned short* Vp = Vtb + (size_t)bh * 64 * SP_;

  int qrow = qt * 16 + l16;
  Frag aq0, aq1;                              // Q as A operand (head dim 64 = 2 k-steps)
  aq0.u[0] = *(const uint4*)(const void*)(Qp + (size_t)qrow * 64 + hi * 16);
  aq0.u[1] = *(const uint4*)(const void*)(Qp + (size_t)qrow * 64 + 8 + hi * 16);
  aq1.u[0] = *(const uint4*)(const void*)(Qp + (size_t)qrow * 64 + 32 + hi * 16);
  aq1.u[1] = *(const uint4*)(const void*)(Qp + (size_t)qrow * 64 + 40 + hi * 16);

  v8f zero = {0.f,0.f,0.f,0.f,0.f,0.f,0.f,0.f};
  v8f o[3] = {zero, zero, zero};              // O^T accumulator (48 dims x 16 queries)
  float m_i = -1e30f, l_i = 0.f;
  const float* myrow = &sS[warp][l16][0];

  for (int kc = 0; kc < SP_ / 32; ++kc) {
    // ---- scores: S[16q x 32k] = Q @ K^T, two 16x16 tiles ----
#pragma unroll
    for (int t = 0; t < 2; ++t) {
      int key = kc * 32 + t * 16 + l16;
      const unsigned short* kr = Kp + (size_t)key * 64;
      Frag bk0, bk1;
      bk0.u[0] = ((const uint4*)(const void*)(kr + hi * 16))[0];
      bk0.u[1] = ((const uint4*)(const void*)(kr + hi * 16))[1];
      bk1.u[0] = ((const uint4*)(const void*)(kr + 32 + hi * 16))[0];
      bk1.u[1] = ((const uint4*)(const void*)(kr + 32 + hi * 16))[1];
      v8f st = zero;
      st = wmma_bf16(aq0.v, bk0.v, st);
      st = wmma_bf16(aq1.v, bk1.v, st);
      bool pad = (key >= S_);
#pragma unroll
      for (int r = 0; r < 8; ++r) {
        float sv = pad ? -1e30f : st[r];
        sS[warp][r + hi * 8][t * 16 + l16] = sv;   // wave-private; in-order DS
      }
    }
    // ---- streaming softmax for query q = l16 ----
    float p[32];
    float cmax = -1e30f;
#pragma unroll
    for (int j4 = 0; j4 < 8; ++j4) {
      float4 f = ((const float4*)(const void*)myrow)[j4];
      p[4*j4+0]=f.x; p[4*j4+1]=f.y; p[4*j4+2]=f.z; p[4*j4+3]=f.w;
      cmax = fmaxf(cmax, fmaxf(fmaxf(f.x,f.y), fmaxf(f.z,f.w)));
    }
    float m_new = fmaxf(m_i, cmax);
    float alpha = __expf(m_i - m_new);
    float psum = 0.f;
#pragma unroll
    for (int j = 0; j < 32; ++j) { p[j] = __expf(p[j] - m_new); psum += p[j]; }
    l_i = l_i * alpha + psum;
    m_i = m_new;
    Frag bp;                                   // P^T as B operand: slot j -> key j+16*hi
#pragma unroll
    for (int j = 0; j < 16; ++j) bp.s[j] = f2bf(p[j + hi * 16]);
#pragma unroll
    for (int t = 0; t < 3; ++t) {
#pragma unroll
      for (int r = 0; r < 8; ++r) o[t][r] *= alpha;   // per-lane (query) rescale
    }
    // ---- O^T += V^T @ P^T : 3 dim-tiles of 16 ----
#pragma unroll
    for (int t = 0; t < 3; ++t) {
      const unsigned short* vr = Vp + (size_t)(t * 16 + l16) * SP_ + kc * 32;
      Frag av;
      av.u[0] = *(const uint4*)(const void*)(vr + hi * 16);
      av.u[1] = *(const uint4*)(const void*)(vr + 8 + hi * 16);
      o[t] = wmma_bf16(av.v, bp.v, o[t]);
    }
  }

  float rl = 1.f / l_i;
  int b = bh / NH_, h = bh % NH_;
  int q = qt * 16 + l16;
  size_t rbase = (size_t)(b * SP_ + q) * D_ + h * HD_;
#pragma unroll
  for (int t = 0; t < 3; ++t)
#pragma unroll
    for (int r = 0; r < 8; ++r)
      Ob[rbase + t * 16 + r + hi * 8] = f2bf(o[t][r] * rl);
}

// ---------------- LayerNorm: wave per row, write fp32 + bf16 ----------------
__global__ __launch_bounds__(256) void ln_kernel(const float* __restrict__ T,
                                                 const float* __restrict__ g,
                                                 const float* __restrict__ be,
                                                 float* __restrict__ X,
                                                 unsigned short* __restrict__ Xb) {
  int row = blockIdx.x * 8 + (threadIdx.x >> 5);
  int lane = threadIdx.x & 31;
  const float* rp = T + (size_t)row * D_;
  float v[6], s = 0.f, sq = 0.f;
#pragma unroll
  for (int i = 0; i < 6; ++i) { v[i] = rp[lane + 32 * i]; s += v[i]; sq += v[i] * v[i]; }
#pragma unroll
  for (int m = 16; m >= 1; m >>= 1) { s += __shfl_xor(s, m, 32); sq += __shfl_xor(sq, m, 32); }
  float mean = s * (1.f / D_);
  float var = sq * (1.f / D_) - mean * mean;
  float rstd = rsqrtf(var + 1e-5f);
#pragma unroll
  for (int i = 0; i < 6; ++i) {
    int c = lane + 32 * i;
    float y = (v[i] - mean) * rstd * g[c] + be[c];
    X[(size_t)row * D_ + c] = y;
    Xb[(size_t)row * D_ + c] = f2bf(y);
  }
}

// ---------------- output: drop id token, drop padding ----------------
__global__ void copy_out_kernel(const float* __restrict__ X, float* __restrict__ out) {
  int idx = blockIdx.x * 256 + threadIdx.x;   // over B*V*D
  int d = idx % D_;
  int vv = (idx / D_) % V_;
  int b = idx / (D_ * V_);
  out[idx] = X[(size_t)(b * SP_ + 1 + vv) * D_ + d];
}

// ---------------- host launcher ----------------
extern "C" void kernel_launch(void* const* d_in, const int* in_sizes, int n_in,
                              void* d_out, int out_size, void* d_ws, size_t ws_size,
                              hipStream_t stream) {
  (void)in_sizes; (void)n_in; (void)out_size;
  const int*   cat_arr    = (const int*)  d_in[0];
  const int*   dt_arr     = (const int*)  d_in[1];
  const int*   amount_arr = (const int*)  d_in[2];
  const int*   id_arr     = (const int*)  d_in[3];
  const float* id_tab     = (const float*)d_in[4];
  const float* cat_tab    = (const float*)d_in[5];
  const float* amount_tab = (const float*)d_in[6];
  const float* dt_tab     = (const float*)d_in[7];
  const float* pos_tab    = (const float*)d_in[8];
  const float* Wqkv       = (const float*)d_in[9];
  const float* bqkv       = (const float*)d_in[10];
  const float* Wo         = (const float*)d_in[11];
  const float* bo         = (const float*)d_in[12];
  const float* W1         = (const float*)d_in[13];
  const float* b1         = (const float*)d_in[14];
  const float* W2         = (const float*)d_in[15];
  const float* b2         = (const float*)d_in[16];
  const float* ln1_g      = (const float*)d_in[17];
  const float* ln1_b      = (const float*)d_in[18];
  const float* ln2_g      = (const float*)d_in[19];
  const float* ln2_b      = (const float*)d_in[20];
  float* out = (float*)d_out;

  // workspace carve-up (~148 MB)
  char* base = (char*)d_ws;
  size_t p = 0;
  auto alloc = [&](size_t bytes) { size_t o = p; p = (p + bytes + 255) & ~(size_t)255; return o; };
  const size_t nXD = (size_t)ROWS_ * D_;                   // 6,684,672
  const size_t nQK = (size_t)B_ * NH_ * SP_ * 64;          // 8,912,896
  float*          X    = (float*)         (base + alloc(nXD * 4));
  unsigned short* Xb   = (unsigned short*)(base + alloc(nXD * 2));
  unsigned short* Qb   = (unsigned short*)(base + alloc(nQK * 2));
  unsigned short* Kb   = (unsigned short*)(base + alloc(nQK * 2));
  unsigned short* Vtb  = (unsigned short*)(base + alloc(nQK * 2));
  unsigned short* Ob   = (unsigned short*)(base + alloc(nXD * 2));
  unsigned short* Hb   = (unsigned short*)(base + alloc(nXD * 2));
  float*          T    = (float*)         (base + alloc(nXD * 4));
  unsigned short* WqkvB= (unsigned short*)(base + alloc((size_t)NL_ * 3 * D_ * D_ * 2));
  unsigned short* WoB  = (unsigned short*)(base + alloc((size_t)NL_ * D_ * D_ * 2));
  unsigned short* W1B  = (unsigned short*)(base + alloc((size_t)NL_ * D_ * D_ * 2));
  unsigned short* W2B  = (unsigned short*)(base + alloc((size_t)NL_ * D_ * D_ * 2));
  (void)ws_size;

  // weights -> bf16
  {
    int n = NL_ * 3 * D_ * D_;
    f2bf_kernel<<<(n + 255) / 256, 256, 0, stream>>>(Wqkv, WqkvB, n);
    n = NL_ * D_ * D_;
    f2bf_kernel<<<(n + 255) / 256, 256, 0, stream>>>(Wo, WoB, n);
    f2bf_kernel<<<(n + 255) / 256, 256, 0, stream>>>(W1, W1B, n);
    f2bf_kernel<<<(n + 255) / 256, 256, 0, stream>>>(W2, W2B, n);
  }

  // embeddings
  init_x_kernel<<<(int)(nXD / 256), 256, 0, stream>>>(id_arr, id_tab, cat_tab, X);
  scatter_kernel<<<B_ * L_, E_, 0, stream>>>(cat_arr, dt_arr, amount_arr, dt_tab, pos_tab, amount_tab, X);
  f2bf_kernel<<<(int)(nXD / 256), 256, 0, stream>>>(X, Xb, (int)nXD);

  const int gQKV = (ROWS_ / 16) * (576 / 64) / 8;   // 2448 blocks
  const int gD   = (ROWS_ / 16) * (192 / 64) / 8;   // 816 blocks
  const int gAtt = (B_ * NH_ * (SP_ / 16)) / 8;     // 1088 blocks
  const int gLN  = ROWS_ / 8;                        // 4352 blocks

  for (int i = 0; i < NL_; ++i) {
    // zero head-dim padding (48..63) of Q/K by clearing whole buffers, then overwrite
    hipMemsetAsync(Qb, 0, nQK * 2, stream);
    hipMemsetAsync(Kb, 0, nQK * 2, stream);
    gemm_kernel<EPI_QKV, 576><<<gQKV, 256, 0, stream>>>(
        Xb, WqkvB + (size_t)i * 3 * D_ * D_, bqkv + i * 3 * D_,
        nullptr, nullptr, nullptr, Qb, Kb, Vtb);
    attn_kernel<<<gAtt, 256, 0, stream>>>(Qb, Kb, Vtb, Ob);
    gemm_kernel<EPI_RES, 192><<<gD, 256, 0, stream>>>(
        Ob, WoB + (size_t)i * D_ * D_, bo + i * D_, X, T, nullptr, nullptr, nullptr, nullptr);
    ln_kernel<<<gLN, 256, 0, stream>>>(T, ln1_g + i * D_, ln1_b + i * D_, X, Xb);
    gemm_kernel<EPI_RELU_BF, 192><<<gD, 256, 0, stream>>>(
        Xb, W1B + (size_t)i * D_ * D_, b1 + i * D_, nullptr, nullptr, Hb, nullptr, nullptr, nullptr);
    gemm_kernel<EPI_RES, 192><<<gD, 256, 0, stream>>>(
        Hb, W2B + (size_t)i * D_ * D_, b2 + i * D_, X, T, nullptr, nullptr, nullptr, nullptr);
    ln_kernel<<<gLN, 256, 0, stream>>>(T, ln2_g + i * D_, ln2_b + i * D_, X, Xb);
  }

  copy_out_kernel<<<(B_ * V_ * D_) / 256, 256, 0, stream>>>(X, out);
}
